// AttentionBlock_39960375722657
// MI455X (gfx1250) — matrix operand order repaired and checked by reading, not compiled, not run on hardware
//
#include <hip/hip_runtime.h>
#include <hip/hip_bf16.h>
#include <stdint.h>

// Problem constants (reference: b=8, c=512, h=w=32, HEADS=8, GROUPS=32)
#define Bb   8
#define Cc   512
#define Nn   1024
#define Hh   8
#define HDd  64

typedef __attribute__((ext_vector_type(16))) __bf16 v16bf;
typedef __attribute__((ext_vector_type(8)))  __bf16 v8bf;
typedef __attribute__((ext_vector_type(8)))  float  v8f;

static __device__ __forceinline__ unsigned short f2bfu(float f) {
  union { float f; unsigned int u; } v; v.f = f;
  unsigned int r = v.u + 0x7FFFu + ((v.u >> 16) & 1u);   // round-to-nearest-even
  return (unsigned short)(r >> 16);
}

static __device__ __forceinline__ v16bf mk16(v8bf lo, v8bf hi) {
  return __builtin_shufflevector(lo, hi, 0,1,2,3,4,5,6,7,8,9,10,11,12,13,14,15);
}
static __device__ __forceinline__ v16bf ld16g(const unsigned short* p) {
  return *(const v16bf*)p;
}
static __device__ __forceinline__ v8bf ld8(const unsigned short* p) {
  return *(const v8bf*)p;
}

// D = A(16x32 bf16) * B(32x16 bf16) + C, fp32 accum
static __device__ __forceinline__ v8f wmma_bf16(v16bf a, v16bf b, v8f c) {
  return __builtin_amdgcn_wmma_f32_16x16x32_bf16(false, a, false, b, (short)0, c, false, false);
}

// Half-wave (16-lane group) reductions — match WMMA C-layout row striping on wave32
static __device__ __forceinline__ float rmax16(float v) {
#pragma unroll
  for (int m = 1; m < 16; m <<= 1) v = fmaxf(v, __shfl_xor(v, m, 32));
  return v;
}
static __device__ __forceinline__ float rsum16(float v) {
#pragma unroll
  for (int m = 1; m < 16; m <<= 1) v += __shfl_xor(v, m, 32);
  return v;
}

// ---------------------------------------------------------------------------
// Kernel 0: convert qkv_w (1536x512) and proj_w (512x512) to bf16
// ---------------------------------------------------------------------------
__global__ void cvt_weights(const float* __restrict__ qw, const float* __restrict__ pw,
                            unsigned short* __restrict__ qwb, unsigned short* __restrict__ pwb) {
  int i = blockIdx.x * blockDim.x + threadIdx.x;
  if (i < 3 * Cc * Cc) {
    qwb[i] = f2bfu(qw[i]);
  } else {
    int j = i - 3 * Cc * Cc;
    if (j < Cc * Cc) pwb[j] = f2bfu(pw[j]);
  }
}

// ---------------------------------------------------------------------------
// Kernel 1: GroupNorm, write xn transposed [b][n][c] as bf16
// one workgroup per (b, group): 16 channels x 1024 pixels
// ---------------------------------------------------------------------------
__global__ void gn_kernel(const float* __restrict__ x, const float* __restrict__ gamma,
                          const float* __restrict__ beta, unsigned short* __restrict__ xnT) {
  int bg  = blockIdx.x;                // 0..255
  int b   = bg >> 5;
  int grp = bg & 31;
  int tid = threadIdx.x;
  const float* xb = x + ((size_t)b * Cc + grp * 16) * Nn;

  float s = 0.f, s2 = 0.f;
  for (int i = tid; i < 16 * Nn; i += 256) { float v = xb[i]; s += v; s2 += v * v; }

  __shared__ float red[2][256];
  red[0][tid] = s; red[1][tid] = s2;
  __syncthreads();
  for (int st = 128; st > 0; st >>= 1) {
    if (tid < st) { red[0][tid] += red[0][tid + st]; red[1][tid] += red[1][tid + st]; }
    __syncthreads();
  }
  float mean = red[0][0] * (1.f / 16384.f);
  float var  = red[1][0] * (1.f / 16384.f) - mean * mean;
  float rs   = rsqrtf(var + 1e-6f);

  for (int i = tid; i < 16 * Nn; i += 256) {
    int cc = i >> 10;            // channel within group
    int n  = i & (Nn - 1);
    int c  = grp * 16 + cc;
    float v = (xb[i] - mean) * rs * gamma[c] + beta[c];
    xnT[((size_t)b * Nn + n) * Cc + c] = f2bfu(v);
  }
}

// ---------------------------------------------------------------------------
// Kernel 2: QKV GEMM.  qkv[b,o,n] = sum_c W[o,c]*xn[b,c,n] + bias[o]
// Each wave: one 16(o) x 64(n) output region, K=512 in steps of 32,
// software-pipelined (register double-buffered fragments).
// Q (scaled by 1/8) and K stored [b][h][n][hd]; V stored [b][h][hd][n].
// ---------------------------------------------------------------------------
__global__ void qkv_gemm(const unsigned short* __restrict__ qwb, const float* __restrict__ bias,
                         const unsigned short* __restrict__ xnT,
                         unsigned short* __restrict__ q_nc, unsigned short* __restrict__ k_nc,
                         unsigned short* __restrict__ v_cn) {
  int wave = (blockIdx.x * blockDim.x + threadIdx.x) >> 5;
  int lane = threadIdx.x & 31;
  int hi = lane >> 4, li = lane & 15;

  int nb = wave & 15;             // 16 n-blocks of 64
  int ot = (wave >> 4) % 96;      // 96 o-tiles of 16
  int b  = wave / (16 * 96);
  int o0 = ot * 16, n0 = nb * 64;

  const unsigned short* wrow = qwb + (size_t)(o0 + li) * Cc;      // A row = o0+li
  const unsigned short* xb   = xnT + (size_t)b * Nn * Cc;

  auto loadA = [&](int kb) {
    return ld16g(wrow + kb + hi * 16);
  };
  auto loadB = [&](int kb, int t) {
    return ld16g(xb + (size_t)(n0 + t * 16 + li) * Cc + kb + hi * 16);
  };

  v8f z = {0.f,0.f,0.f,0.f,0.f,0.f,0.f,0.f};
  v8f acc[4] = {z, z, z, z};

  v16bf a_cur = loadA(0);
  v16bf b_cur[4] = {loadB(0,0), loadB(0,1), loadB(0,2), loadB(0,3)};
#pragma unroll 2
  for (int kb = 0; kb < Cc - 32; kb += 32) {
    v16bf a_nxt = loadA(kb + 32);
    v16bf b_nxt[4] = {loadB(kb+32,0), loadB(kb+32,1), loadB(kb+32,2), loadB(kb+32,3)};
#pragma unroll
    for (int t = 0; t < 4; ++t) acc[t] = wmma_bf16(a_cur, b_cur[t], acc[t]);
    a_cur = a_nxt;
#pragma unroll
    for (int t = 0; t < 4; ++t) b_cur[t] = b_nxt[t];
  }
#pragma unroll
  for (int t = 0; t < 4; ++t) acc[t] = wmma_bf16(a_cur, b_cur[t], acc[t]);

  int s  = o0 / Cc;               // 0=q 1=k 2=v (tile never straddles)
  int oh = (o0 % Cc) / HDd;       // head
  size_t headbase = (size_t)(b * Hh + oh) * Nn * HDd;
#pragma unroll
  for (int r = 0; r < 8; ++r) {
    int o  = o0 + r + hi * 8;
    int cp = o % HDd;
    float bo = bias[o];
#pragma unroll
    for (int t = 0; t < 4; ++t) {
      int n = n0 + t * 16 + li;
      float val = acc[t][r] + bo;
      if (s == 0)      q_nc[headbase + (size_t)n * HDd + cp] = f2bfu(val * 0.125f); // fold 1/sqrt(hd)
      else if (s == 1) k_nc[headbase + (size_t)n * HDd + cp] = f2bfu(val);
      else             v_cn[(size_t)(b * Hh + oh) * HDd * Nn + (size_t)cp * Nn + n] = f2bfu(val);
    }
  }
}

// ---------------------------------------------------------------------------
// Kernel 3: fused flash attention per (b,h). Wave = 16 query rows, stream m
// in chunks of 32 with online softmax; P round-trips LDS (C-layout -> A-frag).
// K fragments double-buffered across chunks; V fragments loaded at chunk top
// so their latency hides under S-WMMAs + softmax VALU + P staging.
// Output written transposed [b][n][c] bf16 for the projection GEMM.
// ---------------------------------------------------------------------------
__global__ void attn_kernel(const unsigned short* __restrict__ q_nc,
                            const unsigned short* __restrict__ k_nc,
                            const unsigned short* __restrict__ v_cn,
                            unsigned short* __restrict__ aoT) {
  int bid  = blockIdx.x;          // 512 blocks
  int nblk = bid & 7;             // 8 n-blocks of 128
  int bh   = bid >> 3;            // b*8+h
  int wv   = threadIdx.x >> 5;    // 8 waves
  int lane = threadIdx.x & 31;
  int hi = lane >> 4, li = lane & 15;
  int n0 = nblk * 128 + wv * 16;

  const unsigned short* Q = q_nc + (size_t)bh * Nn * HDd;
  const unsigned short* K = k_nc + (size_t)bh * Nn * HDd;
  const unsigned short* V = v_cn + (size_t)bh * HDd * Nn;

  // Preload Q^T A-fragments (rows = n, K = c): c 0..31 and 32..63
  v16bf aq[2];
#pragma unroll
  for (int cc = 0; cc < 2; ++cc) {
    const unsigned short* qp = Q + (size_t)(n0 + li) * HDd + cc * 32 + hi * 16;
    aq[cc] = ld16g(qp);
  }

  __shared__ __align__(32) unsigned short pstage[8][16 * 32];
  unsigned short* pw = pstage[wv];

  v8f z = {0.f,0.f,0.f,0.f,0.f,0.f,0.f,0.f};
  v8f O[4] = {z, z, z, z};
  float runmax[8], runsum[8];
#pragma unroll
  for (int r = 0; r < 8; ++r) { runmax[r] = -1e30f; runsum[r] = 0.f; }

  auto loadK = [&](int mbase, v16bf (&bk)[2][2]) {
#pragma unroll
    for (int sub = 0; sub < 2; ++sub)
#pragma unroll
      for (int cc = 0; cc < 2; ++cc)
        bk[sub][cc] = ld16g(K + (size_t)(mbase + sub * 16 + li) * HDd + cc * 32 + hi * 16);
  };

  auto process = [&](int mb, v16bf (&bk)[2][2]) {
    // V fragments for this chunk — independent of P, issue early
    v16bf bv[4];
#pragma unroll
    for (int ct = 0; ct < 4; ++ct)
      bv[ct] = ld16g(V + (size_t)(ct * 16 + li) * Nn + mb + hi * 16);

    // S = (Q^T K): 16 rows x 32 cols (two 16-wide subtiles), scale folded in Q
    v8f sc[2] = {z, z};
#pragma unroll
    for (int sub = 0; sub < 2; ++sub)
#pragma unroll
      for (int cc = 0; cc < 2; ++cc)
        sc[sub] = wmma_bf16(aq[cc], bk[sub][cc], sc[sub]);

    // Online softmax update (rows striped: VGPR r <-> row r + hi*8)
    float f[8];
#pragma unroll
    for (int r = 0; r < 8; ++r) {
      float cm = rmax16(fmaxf(sc[0][r], sc[1][r]));
      float nm = fmaxf(runmax[r], cm);
      f[r] = __expf(runmax[r] - nm);
      runmax[r] = nm;
      float p0 = __expf(sc[0][r] - nm);
      float p1 = __expf(sc[1][r] - nm);
      sc[0][r] = p0; sc[1][r] = p1;
      float cs = rsum16(p0 + p1);
      runsum[r] = runsum[r] * f[r] + cs;
    }
#pragma unroll
    for (int t = 0; t < 4; ++t)
#pragma unroll
      for (int r = 0; r < 8; ++r) O[t][r] *= f[r];

    // P (C-layout) -> LDS (row-major 16x32 bf16) -> A-fragment
#pragma unroll
    for (int r = 0; r < 8; ++r) {
      int row = r + hi * 8;
      pw[row * 32 + li]      = f2bfu(sc[0][r]);
      pw[row * 32 + 16 + li] = f2bfu(sc[1][r]);
    }
    const unsigned short* pr = pw + li * 32 + hi * 16;
    v16bf ap = ld16g(pr);   // per-wave region; DS ops are in-order

    // O += P * V^T   (K = m chunk of 32, N = c in four 16-tiles)
#pragma unroll
    for (int ct = 0; ct < 4; ++ct)
      O[ct] = wmma_bf16(ap, bv[ct], O[ct]);
  };

  // Pipelined m loop: two chunks per iteration, K fragments ping-pong buffered
  v16bf bkA[2][2], bkB[2][2];
  loadK(0, bkA);
  for (int mb = 0; mb < Nn; mb += 64) {
    loadK(mb + 32, bkB);            // mb+32 <= Nn-32, always in range
    process(mb, bkA);
    if (mb + 64 < Nn) loadK(mb + 64, bkA);
    process(mb + 32, bkB);
  }

  // Normalize and store transposed: aoT[b][n][c], c = h*64 + ct*16 + li
  int b = bh >> 3, h = bh & 7;
#pragma unroll
  for (int r = 0; r < 8; ++r) {
    float inv = 1.f / runsum[r];
    int n = n0 + r + hi * 8;
#pragma unroll
    for (int ct = 0; ct < 4; ++ct) {
      int c = h * HDd + ct * 16 + li;
      aoT[((size_t)b * Nn + n) * Cc + c] = f2bfu(O[ct][r] * inv);
    }
  }
}

// ---------------------------------------------------------------------------
// Kernel 4: projection GEMM + bias + residual: out = proj_w @ ao + proj_b + x
// Software-pipelined like the QKV GEMM.
// ---------------------------------------------------------------------------
__global__ void proj_gemm(const unsigned short* __restrict__ pwb, const float* __restrict__ bias,
                          const unsigned short* __restrict__ aoT, const float* __restrict__ x,
                          float* __restrict__ out) {
  int wave = (blockIdx.x * blockDim.x + threadIdx.x) >> 5;
  int lane = threadIdx.x & 31;
  int hi = lane >> 4, li = lane & 15;

  int nb = wave & 15;             // 16 n-blocks of 64
  int ot = (wave >> 4) & 31;      // 32 o-tiles of 16
  int b  = wave >> 9;
  int o0 = ot * 16, n0 = nb * 64;

  const unsigned short* wrow = pwb + (size_t)(o0 + li) * Cc;
  const unsigned short* ab   = aoT + (size_t)b * Nn * Cc;

  auto loadA = [&](int kb) {
    return ld16g(wrow + kb + hi * 16);
  };
  auto loadB = [&](int kb, int t) {
    return ld16g(ab + (size_t)(n0 + t * 16 + li) * Cc + kb + hi * 16);
  };

  v8f z = {0.f,0.f,0.f,0.f,0.f,0.f,0.f,0.f};
  v8f acc[4] = {z, z, z, z};

  v16bf a_cur = loadA(0);
  v16bf b_cur[4] = {loadB(0,0), loadB(0,1), loadB(0,2), loadB(0,3)};
#pragma unroll 2
  for (int kb = 0; kb < Cc - 32; kb += 32) {
    v16bf a_nxt = loadA(kb + 32);
    v16bf b_nxt[4] = {loadB(kb+32,0), loadB(kb+32,1), loadB(kb+32,2), loadB(kb+32,3)};
#pragma unroll
    for (int t = 0; t < 4; ++t) acc[t] = wmma_bf16(a_cur, b_cur[t], acc[t]);
    a_cur = a_nxt;
#pragma unroll
    for (int t = 0; t < 4; ++t) b_cur[t] = b_nxt[t];
  }
#pragma unroll
  for (int t = 0; t < 4; ++t) acc[t] = wmma_bf16(a_cur, b_cur[t], acc[t]);

#pragma unroll
  for (int r = 0; r < 8; ++r) {
    int o = o0 + r + hi * 8;
    float bo = bias[o];
#pragma unroll
    for (int t = 0; t < 4; ++t) {
      int n = n0 + t * 16 + li;
      size_t idx = ((size_t)b * Cc + o) * Nn + n;
      out[idx] = acc[t][r] + bo + x[idx];
    }
  }
}

// ---------------------------------------------------------------------------
extern "C" void kernel_launch(void* const* d_in, const int* in_sizes, int n_in,
                              void* d_out, int out_size, void* d_ws, size_t ws_size,
                              hipStream_t stream) {
  const float* x      = (const float*)d_in[0];
  const float* norm_w = (const float*)d_in[1];
  const float* norm_b = (const float*)d_in[2];
  const float* qkv_w  = (const float*)d_in[3];
  const float* qkv_b  = (const float*)d_in[4];
  const float* proj_w = (const float*)d_in[5];
  const float* proj_b = (const float*)d_in[6];
  float* out = (float*)d_out;

  unsigned short* ws  = (unsigned short*)d_ws;
  unsigned short* xnT = ws;                         // 8*1024*512      = 4,194,304
  unsigned short* qwb = xnT + (size_t)Bb * Nn * Cc; // 1536*512        =   786,432
  unsigned short* pwb = qwb + (size_t)3 * Cc * Cc;  // 512*512         =   262,144
  unsigned short* qnc = pwb + (size_t)Cc * Cc;      // 8*8*1024*64     = 4,194,304
  unsigned short* knc = qnc + (size_t)Bb * Hh * Nn * HDd;
  unsigned short* vcn = knc + (size_t)Bb * Hh * Nn * HDd;
  unsigned short* aoT = vcn + (size_t)Bb * Hh * HDd * Nn;
  // total: 22,020,096 ushorts = 42 MiB of workspace

  cvt_weights<<<4096, 256, 0, stream>>>(qkv_w, proj_w, qwb, pwb);
  gn_kernel  <<<Bb * 32, 256, 0, stream>>>(x, norm_w, norm_b, xnT);
  qkv_gemm   <<<1536, 256, 0, stream>>>(qwb, qkv_b, xnT, qnc, knc, vcn);
  attn_kernel<<<512, 256, 0, stream>>>(qnc, knc, vcn, aoT);
  proj_gemm  <<<512, 256, 0, stream>>>(pwb, proj_b, aoT, x, out);
}
